// DWT_Features_69647189672625
// MI455X (gfx1250) — matrix operand order repaired and checked by reading, not compiled, not run on hardware
//
#include <hip/hip_runtime.h>

// ---------------------------------------------------------------------------
// DWT features for MI455X (gfx1250, wave32).
//
// feat = leakyrelu( X_s @ Weff_s^T + bias ),  Weff = DWT matrix folded into
// conv weights.  Main GEMM is memory-bound (128 MB of X @ 23.3 TB/s ~ 5.5us),
// done with v_wmma_f32_16x16x32_f16 (f16 in, f32 accumulate).  B tiles are
// staged into double-buffered LDS -- via GLOBAL_LOAD_ASYNC_TO_LDS_B128
// (ASYNCcnt-tracked DMA, no VGPR round-trip) when the toolchain exposes the
// builtin, else via the proven global->reg->ds_store path.  X is streamed
// from HBM with global_prefetch.
// ---------------------------------------------------------------------------

typedef __attribute__((ext_vector_type(16))) _Float16 v16h;
typedef __attribute__((ext_vector_type(8)))  float    v8f;
typedef __attribute__((ext_vector_type(4)))  int      v4i;

#define NB      2048          // batch
#define NSW     4             // subwindows
#define NK      128           // conv out channels per subwindow
#define KDIM    4096          // 64 taps * 64 (h*w) reduction per subwindow
#define NSTEP   128           // KDIM / 32 (WMMA K = 32)
#define XROW    16384         // floats per batch row of x (256*8*8)

#if __has_builtin(__builtin_amdgcn_global_load_async_to_lds_b128) && \
    __has_builtin(__builtin_amdgcn_s_wait_asynccnt)
#define CDNA5_ASYNC 1
#else
#define CDNA5_ASYNC 0
#endif

#if CDNA5_ASYNC
// Builtin signature (from the clang diagnostic): pointers are b128-typed,
// global side is addrspace(1) ("__device__"), LDS side addrspace(3).
// Per CDNA5 ISA the same INST_OFFSET is added to both the
// global and the LDS address, so one (gaddr, ldsaddr) pair + imm offsets
// moves a whole 64B/thread slice.
#define ASYNC_B128(g, l, off)                                              \
    __builtin_amdgcn_global_load_async_to_lds_b128(                        \
        (__attribute__((address_space(1))) v4i*)(g),                       \
        (__attribute__((address_space(3))) v4i*)(l), (off), 0)
#endif

// db4 analysis filters, REVERSED (reference uses correlation with reversed
// filters == plain convolution with originals).
__device__ __constant__ float c_hlo[8] = {
    0.23037781330885523f,  0.7148465705525415f,   0.6308807679295904f,
   -0.02798376941698385f, -0.18703481171888114f,  0.030841381835986965f,
    0.032883011666982945f, -0.010597401784997278f };
__device__ __constant__ float c_hhi[8] = {
   -0.010597401784997278f, -0.032883011666982945f, 0.030841381835986965f,
    0.18703481171888114f,  -0.02798376941698385f,  -0.6308807679295904f,
    0.7148465705525415f,   -0.23037781330885523f };

// ---------------------------------------------------------------------------
// Phase A: build the 84x64 DWT analysis matrix D (coeffs = D @ x_time) by
// pushing the 64 identity basis vectors through the 3-level reflect-pad DWT.
// Thread tau computes column tau.  D stored row-major [84][64].
// ---------------------------------------------------------------------------
__global__ void build_dwt_matrix(float* __restrict__ D)
{
    const int tau = threadIdx.x;          // 0..63
    float cur[64];
    float out[84];
    for (int i = 0; i < 64; ++i) cur[i] = (i == tau) ? 1.f : 0.f;

    const int OS[3] = {35, 21, 14};       // outsizes per level (N: 64->35->21->14)
    const int HO[3] = {14, 49, 70};       // hi-band offsets in concat [lo3|hi1|hi2|hi3]

    int N = 64;
    for (int lev = 0; lev < 3; ++lev) {
        const int osz = OS[lev];
        const int p   = 2 * (osz - 1) - N + 8;
        const int Nz  = N + (p & 1);      // odd total pad: one zero appended first
        const int pad = p >> 1;
        float lo[35];
        for (int o = 0; o < osz; ++o) {
            float aL = 0.f, aH = 0.f;
#pragma unroll
            for (int l = 0; l < 8; ++l) {
                int i = 2 * o + l - pad;          // index into (zero-ext) signal
                int j = (i < 0) ? -i : i;          // numpy-style 'reflect'
                if (j >= Nz) j = 2 * (Nz - 1) - j;
                const float xv = (j >= N) ? 0.f : cur[j];
                aL += xv * c_hlo[l];
                aH += xv * c_hhi[l];
            }
            lo[o] = aL;
            out[HO[lev] + o] = aH;
        }
        for (int o = 0; o < osz; ++o) cur[o] = lo[o];
        N = osz;
    }
    for (int o = 0; o < 14; ++o) out[o] = cur[o];   // final low-pass band

    for (int t = 0; t < 84; ++t) D[t * 64 + tau] = out[t];
}

// ---------------------------------------------------------------------------
// Phase B: fold D into conv weights and emit f16 weights PRE-SWIZZLED into the
// WMMA B-operand layout:
//   Wsw chunk index (((s*128 + c)*8 + j)*32 + lane) holds 16 contiguous halfs
//     = Weff[s][n = j*16 + lane%16][k = c*32 + (lane/16)*16 + 0..15]
// Block = one (s,n); thread = (tau-quarter, hw); D cached in LDS; 16 f32
// accumulators per thread so each conv_weight element is loaded once/16 FMA.
// ---------------------------------------------------------------------------
__global__ __launch_bounds__(256) void fold_weights(const float* __restrict__ wt,
                                                    const float* __restrict__ D,
                                                    _Float16* __restrict__ Wsw)
{
    __shared__ __align__(16) float Dlds[84 * 64];
    for (int i = threadIdx.x; i < 84 * 64; i += 256) Dlds[i] = D[i];
    __syncthreads();

    const int sn   = blockIdx.x;          // 0..511
    const int s    = sn >> 7;
    const int n    = sn & 127;
    const int hw   = threadIdx.x & 63;    // 0..63
    const int tq   = threadIdx.x >> 6;    // 0..3
    const int tau0 = tq * 16;

    const float* wp = wt + (size_t)(s * 128 + n) * 84 * 64 + hw;

    float acc[16];
#pragma unroll
    for (int i = 0; i < 16; ++i) acc[i] = 0.f;

    for (int t = 0; t < 84; ++t) {
        const float w = wp[t * 64];
        const float4* dp = (const float4*)&Dlds[t * 64 + tau0];
        const float4 d0 = dp[0], d1 = dp[1], d2 = dp[2], d3 = dp[3];
        acc[0]  += w * d0.x;  acc[1]  += w * d0.y;  acc[2]  += w * d0.z;  acc[3]  += w * d0.w;
        acc[4]  += w * d1.x;  acc[5]  += w * d1.y;  acc[6]  += w * d1.z;  acc[7]  += w * d1.w;
        acc[8]  += w * d2.x;  acc[9]  += w * d2.y;  acc[10] += w * d2.z;  acc[11] += w * d2.w;
        acc[12] += w * d3.x;  acc[13] += w * d3.y;  acc[14] += w * d3.z;  acc[15] += w * d3.w;
    }

    const int j        = n >> 4;
    const int lanebase = n & 15;
#pragma unroll
    for (int i = 0; i < 16; ++i) {
        const int tau   = tau0 + i;
        const int k     = tau * 64 + hw;
        const int c     = k >> 5;
        const int khalf = (k >> 4) & 1;
        const int h16   = k & 15;
        const int lane  = khalf * 16 + lanebase;
        const size_t off = (((size_t)(s * 128 + c) * 8 + j) * 32 + lane) * 16 + h16;
        Wsw[off] = (_Float16)acc[i];
    }
}

// ---------------------------------------------------------------------------
// Phase C: main GEMM.  Grid (32, 4): blockIdx.x -> 64 batch rows, blockIdx.y
// -> subwindow s.  4 waves/WG; wave owns 16 rows x all 128 outputs
// (8 wmma accumulators).  B (swizzled weights) double-buffered in LDS.
// ---------------------------------------------------------------------------
__global__ __launch_bounds__(128) void dwt_gemm(const float*    __restrict__ X,
                                                const _Float16* __restrict__ Wsw,
                                                const float*    __restrict__ bias,
                                                float*          __restrict__ Y)
{
    const int s    = blockIdx.y;
    const int wv   = threadIdx.x >> 5;     // wave 0..3
    const int lane = threadIdx.x & 31;
    const int kh   = lane >> 4;            // lane half (K sub-block select)
    const int l16  = lane & 15;

    const int    mrow = blockIdx.x * 64 + wv * 16 + l16;   // A-matrix row M = lane%16
    const float* xrow = X + (size_t)mrow * XROW + s * KDIM;

    __shared__ __align__(32) _Float16 smB[2][8 * 32 * 16]; // [buf][j][lane][16]

    v8f zero = {};
    v8f acc[8];
#pragma unroll
    for (int j = 0; j < 8; ++j) acc[j] = zero;

    const _Float16* wbase = Wsw + (size_t)s * 128 * 4096;  // (s, c=0) block

    // ---- Stage step 0 B tile into LDS buffer 0 (8 KB; 64 B per thread). ----
#if CDNA5_ASYNC
    {
        const char* g = (const char*)wbase + (size_t)threadIdx.x * 32;
        char*       l = (char*)&smB[0][0] + (size_t)threadIdx.x * 32;
        ASYNC_B128(g, l, 0);
        ASYNC_B128(g, l, 16);
        ASYNC_B128(g, l, 4096);
        ASYNC_B128(g, l, 4112);
        __builtin_amdgcn_s_wait_asynccnt(0);
    }
#else
    {
        const v16h* g = (const v16h*)wbase;
        v16h r0 = g[threadIdx.x];
        v16h r1 = g[threadIdx.x + 128];
        v16h* d = (v16h*)smB[0];
        d[threadIdx.x]       = r0;
        d[threadIdx.x + 128] = r1;
    }
#endif
    __syncthreads();

    for (int c = 0; c < NSTEP; ++c) {
#if CDNA5_ASYNC
        // Kick off the DMA of the next B tile; it lands in the other LDS
        // buffer while this step's WMMAs run.  Writing buf (c+1)&1 here is
        // safe: its step-(c-1) readers finished their ds_loads before the
        // barrier that ended step c-1.
        if (c + 1 < NSTEP) {
            const char* g = (const char*)(wbase + (size_t)(c + 1) * 4096)
                            + (size_t)threadIdx.x * 32;
            char*       l = (char*)&smB[(c + 1) & 1][0] + (size_t)threadIdx.x * 32;
            ASYNC_B128(g, l, 0);
            ASYNC_B128(g, l, 16);
            ASYNC_B128(g, l, 4096);
            ASYNC_B128(g, l, 4112);
        }
#else
        // Prefetch next B tile into registers (hits WGP$/L2: weights are hot).
        v16h r0 = {}, r1 = {};
        if (c + 1 < NSTEP) {
            const v16h* g = (const v16h*)(wbase + (size_t)(c + 1) * 4096);
            r0 = g[threadIdx.x];
            r1 = g[threadIdx.x + 128];
        }
#endif
        // Prefetch X stream a few steps ahead (global_prefetch_b8).
        if (c + 8 < NSTEP)
            __builtin_prefetch(xrow + (c + 8) * 32 + 8 * kh, 0, 1);

        // Load A operand: lane needs X[mrow][c*32 + 16*kh + 0..15] in halfs 0..15
        // (contiguous k order, matching the swizzled B layout).
        const float* xp = xrow + c * 32;
        const float4 p0 = *(const float4*)(xp + 16 * kh);
        const float4 p1 = *(const float4*)(xp + 16 * kh + 4);
        const float4 q0 = *(const float4*)(xp + 16 * kh + 8);
        const float4 q1 = *(const float4*)(xp + 16 * kh + 12);
        v16h a;
        a[0]  = (_Float16)p0.x; a[1]  = (_Float16)p0.y; a[2]  = (_Float16)p0.z; a[3]  = (_Float16)p0.w;
        a[4]  = (_Float16)p1.x; a[5]  = (_Float16)p1.y; a[6]  = (_Float16)p1.z; a[7]  = (_Float16)p1.w;
        a[8]  = (_Float16)q0.x; a[9]  = (_Float16)q0.y; a[10] = (_Float16)q0.z; a[11] = (_Float16)q0.w;
        a[12] = (_Float16)q1.x; a[13] = (_Float16)q1.y; a[14] = (_Float16)q1.z; a[15] = (_Float16)q1.w;

        const v16h* bb = (const v16h*)smB[c & 1];
#pragma unroll
        for (int j = 0; j < 8; ++j) {
            const v16h b = bb[j * 32 + lane];
            acc[j] = __builtin_amdgcn_wmma_f32_16x16x32_f16(
                         false, a, false, b, (short)0, acc[j], false, false);
        }

        if (c + 1 < NSTEP) {
#if CDNA5_ASYNC
            __builtin_amdgcn_s_wait_asynccnt(0);   // next tile landed in LDS
#else
            v16h* d = (v16h*)smB[(c + 1) & 1];
            d[threadIdx.x]       = r0;
            d[threadIdx.x + 128] = r1;
#endif
            __syncthreads();   // single barrier/step: buffers alternate safely
        }
    }

    // Epilogue: C/D layout -> lane l, vgpr v holds (M = v + 8*(l/16), N = l%16).
#pragma unroll
    for (int j = 0; j < 8; ++j) {
        const int   n  = j * 16 + l16;
        const float bv = bias[s * 128 + n];
#pragma unroll
        for (int v = 0; v < 8; ++v) {
            const int m = blockIdx.x * 64 + wv * 16 + v + 8 * kh;
            float val = acc[j][v] + bv;
            val = (val > 0.f) ? val : 0.01f * val;
            Y[(size_t)m * (NSW * NK) + s * NK + n] = val;
        }
    }
}

// ---------------------------------------------------------------------------
extern "C" void kernel_launch(void* const* d_in, const int* in_sizes, int n_in,
                              void* d_out, int out_size, void* d_ws, size_t ws_size,
                              hipStream_t stream)
{
    (void)in_sizes; (void)n_in; (void)out_size; (void)ws_size;

    const float* x    = (const float*)d_in[0];  // [2048,1,256,8,8] f32
    const float* wt   = (const float*)d_in[1];  // [4,128,84,8,8]   f32
    const float* bias = (const float*)d_in[2];  // [4,128]          f32
    float*       out  = (float*)d_out;          // [2048,512]       f32

    // Workspace: D (84*64 f32 = 21.5 KB) @ 0, swizzled f16 weights (4 MB) @ 32 KB.
    float*    D   = (float*)d_ws;
    _Float16* Wsw = (_Float16*)((char*)d_ws + 32768);

    build_dwt_matrix<<<dim3(1),        dim3(64),  0, stream>>>(D);
    fold_weights    <<<dim3(512),      dim3(256), 0, stream>>>(wt, D, Wsw);
    dwt_gemm        <<<dim3(32, 4),    dim3(128), 0, stream>>>(x, Wsw, bias, out);
}
